// GAT_21303037788172
// MI455X (gfx1250) — compile-verified, matrix operand order checked by reading, not run on hardware
//
#include <hip/hip_runtime.h>
#include <hip/hip_bf16.h>

typedef __attribute__((ext_vector_type(16))) __bf16 v16bf;
typedef __attribute__((ext_vector_type(8)))  __bf16 v8bf;
typedef __attribute__((ext_vector_type(8)))  float  v8f;

#define T_     64
#define NN     2048
#define KN     32
#define IND    256
#define H1_    128
#define H2_    128
#define GG     256
#define PP     60
#define NHEADS 3
#define MTOT   (T_*NN)   // 131072

// ---------------------------------------------------------------------------
// Kernel 0: convert weights to bf16, transposed to N-major (Bt[n][k]) so the
// WMMA B operand loads as two contiguous 16B chunks per lane (same pattern as A).
//   fc1 (3, 256, 128) -> fc1bt[(h*128+n)*256 + k]
//   fc2 (384, 128)    -> fc2bt[n*384 + k]
// ---------------------------------------------------------------------------
__global__ void convert_weights(const float* __restrict__ fc1,
                                const float* __restrict__ fc2,
                                __bf16* __restrict__ fc1bt,
                                __bf16* __restrict__ fc2bt) {
    int i = blockIdx.x * blockDim.x + threadIdx.x;
    const int n1 = NHEADS * H1_ * IND;          // 98304
    const int n2 = H2_ * (NHEADS * H1_);        // 49152
    if (i < n1) {
        int k = i % IND;
        int row = i / IND;                      // h*128+n
        int n = row % H1_, h = row / H1_;
        fc1bt[i] = (__bf16)fc1[((size_t)h * IND + k) * H1_ + n];
    } else if (i < n1 + n2) {
        int j = i - n1;
        int k = j % (NHEADS * H1_);
        int n = j / (NHEADS * H1_);
        fc2bt[j] = (__bf16)fc2[(size_t)k * H2_ + n];
    }
}

// ---------------------------------------------------------------------------
// Kernel 1: input attention. Wave32 per (t,n). Lane = neighbor k.
//   e_k = (dot(src_k, z_n)/|src_k|/|z_n|)^4 ;  ih = sum_k e_k * src_k
// Output bf16 (feeds WMMA GEMM directly).
// ---------------------------------------------------------------------------
__global__ __launch_bounds__(256) void input_attn(const float* __restrict__ h,
                                                  const int* __restrict__ neigh,
                                                  __bf16* __restrict__ ihbf) {
    int wid = blockIdx.x * 8 + (threadIdx.x >> 5);
    int lane = threadIdx.x & 31;
    if (wid >= MTOT) return;
    int t = wid / NN, n = wid % NN;
    const float* dst = h + (size_t)wid * IND;
    int nb = neigh[n * KN + lane];
    const float* src = h + ((size_t)t * NN + nb) * IND;
    float dot = 0.f, ns = 0.f, nd = 0.f;
    for (int d = 0; d < IND; ++d) {
        float s = src[d], z = dst[d];
        dot += s * z; ns += s * s; nd += z * z;
    }
    float c = dot * rsqrtf(ns) * rsqrtf(nd);
    float e = c * c; e *= e;                    // ^4
    float acc[8];
#pragma unroll
    for (int j = 0; j < 8; ++j) acc[j] = 0.f;
    for (int k = 0; k < KN; ++k) {
        float ek = __shfl(e, k, 32);
        int nbk  = __shfl(nb, k, 32);
        const float* sk = h + ((size_t)t * NN + nbk) * IND;
#pragma unroll
        for (int j = 0; j < 8; ++j) acc[j] += ek * sk[lane + 32 * j];
    }
    __bf16* out = ihbf + (size_t)wid * IND;
#pragma unroll
    for (int j = 0; j < 8; ++j) out[lane + 32 * j] = (__bf16)acc[j];
}

// ---------------------------------------------------------------------------
// Kernel 2/4: register-blocked bf16 WMMA GEMM.
//   C(M,N) = A(M,Kd) * Bt(N,Kd)^T, f32 accumulate.
// One wave owns a 16 x (16*NB) strip: NB accumulator fragments, one A fragment
// reused NB times per 32-deep K chunk (8 WMMAs per A load with NB=8).
// Per-lane loads follow the documented 16-bit operand layout:
//   lane L: row = L&15, half = L>>4 ; elems 0..7 -> K = kc+8*half+0..7,
//   elems 8..15 -> K = kc+16+8*half+0..7  (two contiguous 16B loads).
// ---------------------------------------------------------------------------
template <int NB>
__global__ __launch_bounds__(256) void gemm_bf16_blk(const __bf16* __restrict__ A,
                                                     const __bf16* __restrict__ Bt,
                                                     float* __restrict__ C,
                                                     int M, int N, int Kd) {
    int wid = blockIdx.x * 8 + (threadIdx.x >> 5);
    int lane = threadIdx.x & 31;
    int nGroups = N / (16 * NB);
    int total = (M >> 4) * nGroups;
    if (wid >= total) return;
    int mt = wid / nGroups, ng = wid % nGroups;
    int m0 = mt << 4, n0 = ng * 16 * NB;
    int half = lane >> 4, r16 = lane & 15;
    const __bf16* arow  = A  + (size_t)(m0 + r16) * Kd;
    const __bf16* bbase = Bt + (size_t)(n0 + r16) * Kd;
    v8f acc[NB];
#pragma unroll
    for (int nb = 0; nb < NB; ++nb) acc[nb] = v8f{};
    for (int kc = 0; kc < Kd; kc += 32) {
        union { v16bf v; v8bf p[2]; } a;
        a.p[0] = *(const v8bf*)(arow + kc + 8 * half);
        a.p[1] = *(const v8bf*)(arow + kc + 16 + 8 * half);
#pragma unroll
        for (int nb = 0; nb < NB; ++nb) {
            union { v16bf v; v8bf p[2]; } b;
            const __bf16* brow = bbase + (size_t)(nb * 16) * Kd;
            b.p[0] = *(const v8bf*)(brow + kc + 8 * half);
            b.p[1] = *(const v8bf*)(brow + kc + 16 + 8 * half);
            acc[nb] = __builtin_amdgcn_wmma_f32_16x16x32_bf16(
                false, a.v, false, b.v, (short)0, acc[nb], false, false);
        }
    }
#pragma unroll
    for (int nb = 0; nb < NB; ++nb)
#pragma unroll
        for (int r = 0; r < 8; ++r)
            C[(size_t)(m0 + r + 8 * half) * N + n0 + nb * 16 + r16] = acc[nb][r];
}

// ---------------------------------------------------------------------------
// Kernel 3: per-(node,head) GAT softmax attention on z (M x 384), relu, bf16 out.
// ---------------------------------------------------------------------------
__global__ __launch_bounds__(256) void gat_heads(const float* __restrict__ z,
                                                 const int* __restrict__ neigh,
                                                 __bf16* __restrict__ c0) {
    int wid = blockIdx.x * 8 + (threadIdx.x >> 5);
    int lane = threadIdx.x & 31;
    if (wid >= MTOT * NHEADS) return;
    int hd = wid % NHEADS;
    int node = wid / NHEADS;
    int t = node / NN, n = node % NN;
    const float* zn = z + (size_t)node * (NHEADS * H1_) + hd * H1_;
    int nb = neigh[n * KN + lane];
    const float* zs = z + ((size_t)t * NN + nb) * (NHEADS * H1_) + hd * H1_;
    float e = 0.f;
    for (int d = 0; d < H1_; ++d) e += zs[d] * zn[d];
    float m = e;
    for (int o = 16; o > 0; o >>= 1) m = fmaxf(m, __shfl_xor(m, o, 32));
    float p = __expf(e - m);
    float s = p;
    for (int o = 16; o > 0; o >>= 1) s += __shfl_xor(s, o, 32);
    float a = p / s;
    float acc[4] = {0.f, 0.f, 0.f, 0.f};
    for (int k = 0; k < KN; ++k) {
        float ak = __shfl(a, k, 32);
        int nbk  = __shfl(nb, k, 32);
        const float* sk = z + ((size_t)t * NN + nbk) * (NHEADS * H1_) + hd * H1_;
#pragma unroll
        for (int j = 0; j < 4; ++j) acc[j] += ak * sk[lane + 32 * j];
    }
    __bf16* out = c0 + (size_t)node * (NHEADS * H1_) + hd * H1_;
#pragma unroll
    for (int j = 0; j < 4; ++j) out[lane + 32 * j] = (__bf16)fmaxf(acc[j], 0.f);
}

// ---------------------------------------------------------------------------
// Kernel 5: GAT attention on z2 (M x 128), f32 out (relu applied at pooling).
// ---------------------------------------------------------------------------
__global__ __launch_bounds__(256) void gat_out(const float* __restrict__ z2,
                                               const int* __restrict__ neigh,
                                               float* __restrict__ c2) {
    int wid = blockIdx.x * 8 + (threadIdx.x >> 5);
    int lane = threadIdx.x & 31;
    if (wid >= MTOT) return;
    int t = wid / NN, n = wid % NN;
    const float* zn = z2 + (size_t)wid * H2_;
    int nb = neigh[n * KN + lane];
    const float* zs = z2 + ((size_t)t * NN + nb) * H2_;
    float e = 0.f;
    for (int d = 0; d < H2_; ++d) e += zs[d] * zn[d];
    float m = e;
    for (int o = 16; o > 0; o >>= 1) m = fmaxf(m, __shfl_xor(m, o, 32));
    float p = __expf(e - m);
    float s = p;
    for (int o = 16; o > 0; o >>= 1) s += __shfl_xor(s, o, 32);
    float a = p / s;
    float acc[4] = {0.f, 0.f, 0.f, 0.f};
    for (int k = 0; k < KN; ++k) {
        float ak = __shfl(a, k, 32);
        int nbk  = __shfl(nb, k, 32);
        const float* sk = z2 + ((size_t)t * NN + nbk) * H2_;
#pragma unroll
        for (int j = 0; j < 4; ++j) acc[j] += ak * sk[lane + 32 * j];
    }
    float* out = c2 + (size_t)wid * H2_;
#pragma unroll
    for (int j = 0; j < 4; ++j) out[lane + 32 * j] = acc[j];
}

// ---------------------------------------------------------------------------
// Kernel 6: pooled[t][d] = max_n relu(c2[t,n,d]) = max(0, max_n c2)
// ---------------------------------------------------------------------------
__global__ void pool_max(const float* __restrict__ c2, float* __restrict__ pooled) {
    int t = blockIdx.x, d = threadIdx.x;
    float m = 0.f;
    const float* base = c2 + (size_t)t * NN * H2_ + d;
    for (int n = 0; n < NN; ++n) m = fmaxf(m, base[(size_t)n * H2_]);
    pooled[t * H2_ + d] = m;
}

// ---------------------------------------------------------------------------
// Kernel 7: sequential tail — GRU over t, residual heads, SIR scan (P=60).
// Single 256-thread block (work per step is tiny).
// ---------------------------------------------------------------------------
__global__ __launch_bounds__(256) void seq_tail(
    const float* __restrict__ pooled,
    const float* __restrict__ Wih, const float* __restrict__ Whh,
    const float* __restrict__ bih, const float* __restrict__ bhh,
    const float* __restrict__ res1W, const float* __restrict__ res1b,
    const float* __restrict__ res2W, const float* __restrict__ res2b,
    const float* __restrict__ hx0, const float* __restrict__ Npop,
    const float* __restrict__ I, const float* __restrict__ R,
    const float* __restrict__ S, const float* __restrict__ It,
    const float* __restrict__ Rt, float* __restrict__ out) {
    __shared__ float hx[GG], xg[H2_], gi[3 * GG], gh[3 * GG], nhx[GG + 2];
    __shared__ float pred[2 * PP], sab[2];
    int tid = threadIdx.x;
    if (tid < GG) hx[tid] = hx0[tid];
    __syncthreads();
    float Np = Npop[0];
    for (int t = 0; t < T_; ++t) {
        if (tid < H2_) xg[tid] = pooled[t * H2_ + tid];
        __syncthreads();
        for (int i = tid; i < 3 * GG; i += 256) {
            float s1 = bih[i];
            const float* w1 = Wih + (size_t)i * H2_;
            for (int j = 0; j < H2_; ++j) s1 += w1[j] * xg[j];
            gi[i] = s1;
            float s2 = bhh[i];
            const float* w2 = Whh + (size_t)i * GG;
            for (int j = 0; j < GG; ++j) s2 += w2[j] * hx[j];
            gh[i] = s2;
        }
        __syncthreads();
        // GG == blockDim.x == 256: every thread owns one gate lane
        {
            float r  = 1.f / (1.f + __expf(-(gi[tid] + gh[tid])));
            float zg = 1.f / (1.f + __expf(-(gi[GG + tid] + gh[GG + tid])));
            float nn = tanhf(gi[2 * GG + tid] + r * gh[2 * GG + tid]);
            float h2 = (1.f - zg) * nn + zg * hx[tid];
            hx[tid] = h2;          // each thread reads/writes only its own slot
            nhx[tid] = h2;
        }
        if (tid == 0) { nhx[GG] = It[t]; nhx[GG + 1] = Rt[t]; }
        __syncthreads();
        if (tid < 2 * PP) {
            float s = res1b[tid];
            const float* w = res1W + (size_t)tid * (GG + 2);
            for (int j = 0; j < GG + 2; ++j) s += w[j] * nhx[j];
            pred[tid] = s;
        }
        if (tid < 2) {
            float s = res2b[tid];
            const float* w = res2W + (size_t)tid * (GG + 2);
            for (int j = 0; j < GG + 2; ++j) s += w[j] * nhx[j];
            sab[tid] = 1.f / (1.f + __expf(-s));
        }
        __syncthreads();
        if (tid < PP) {
            out[t * PP + tid]            = pred[2 * tid];       // new_I
            out[T_ * PP + t * PP + tid]  = pred[2 * tid + 1];   // new_R
        }
        if (tid == 0) {
            float a_s = sab[0], b_s = sab[1];
            float lI = I[t], lR = R[t], lS = S[t];
            float* pI = out + 2 * T_ * PP + t * PP;
            float* pR = out + 3 * T_ * PP + t * PP;
            for (int p = 0; p < PP; ++p) {
                float dI = a_s * lI * (lS / Np) - b_s * lI;
                float dR = b_s * lI;
                pI[p] = dI; pR[p] = dR;
                lI += dI; lR += dR;
                lS = Np - lI - lR;
            }
        }
        __syncthreads();
    }
}

// ---------------------------------------------------------------------------
extern "C" void kernel_launch(void* const* d_in, const int* in_sizes, int n_in,
                              void* d_out, int out_size, void* d_ws, size_t ws_size,
                              hipStream_t stream) {
    (void)in_sizes; (void)n_in; (void)out_size; (void)ws_size;
    const float* h     = (const float*)d_in[0];
    const int*   neigh = (const int*)  d_in[1];
    const float* Npop  = (const float*)d_in[2];
    const float* I_    = (const float*)d_in[3];
    const float* R_    = (const float*)d_in[4];
    const float* S_    = (const float*)d_in[5];
    const float* It    = (const float*)d_in[6];
    const float* Rt    = (const float*)d_in[7];
    const float* fc1   = (const float*)d_in[8];
    const float* fc2   = (const float*)d_in[9];
    const float* Wih   = (const float*)d_in[10];
    const float* Whh   = (const float*)d_in[11];
    const float* bih   = (const float*)d_in[12];
    const float* bhh   = (const float*)d_in[13];
    const float* r1W   = (const float*)d_in[14];
    const float* r1b   = (const float*)d_in[15];
    const float* r2W   = (const float*)d_in[16];
    const float* r2b   = (const float*)d_in[17];
    const float* hx0   = (const float*)d_in[18];
    float* out = (float*)d_out;

    size_t off = 0;
    char* base = (char*)d_ws;
    auto wsalloc = [&](size_t bytes) -> void* {
        void* p = base + off;
        off += (bytes + 255) & ~(size_t)255;
        return p;
    };
    __bf16* fc1bt = (__bf16*)wsalloc((size_t)NHEADS * H1_ * IND * 2);
    __bf16* fc2bt = (__bf16*)wsalloc((size_t)H2_ * NHEADS * H1_ * 2);
    __bf16* ihbf  = (__bf16*)wsalloc((size_t)MTOT * IND * 2);
    float*  z     = (float*) wsalloc((size_t)MTOT * NHEADS * H1_ * 4);
    __bf16* c0bf  = (__bf16*)wsalloc((size_t)MTOT * NHEADS * H1_ * 2);
    float*  z2    = (float*) wsalloc((size_t)MTOT * H2_ * 4);
    float*  c2    = (float*) wsalloc((size_t)MTOT * H2_ * 4);
    float*  pooled= (float*) wsalloc((size_t)T_ * H2_ * 4);

    // 0) weight conversion (bf16, N-major)
    {
        int total = NHEADS * H1_ * IND + H2_ * NHEADS * H1_;
        convert_weights<<<(total + 255) / 256, 256, 0, stream>>>(fc1, fc2, fc1bt, fc2bt);
    }
    // 1) input attention -> ih (bf16)
    input_attn<<<MTOT / 8, 256, 0, stream>>>(h, neigh, ihbf);
    // 2) z = ih @ fc1 (3 heads concatenated, N=384), WMMA bf16, 16x128 strips
    {
        int waves = (MTOT / 16) * ((NHEADS * H1_) / 128);   // 8192 * 3
        gemm_bf16_blk<8><<<waves / 8, 256, 0, stream>>>(ihbf, fc1bt, z,
                                                        MTOT, NHEADS * H1_, IND);
    }
    // 3) per-head softmax attention + relu -> c0 (bf16, M x 384)
    gat_heads<<<MTOT * NHEADS / 8, 256, 0, stream>>>(z, neigh, c0bf);
    // 4) z2 = c0 @ fc2 (N=128), WMMA bf16, 16x128 strips
    {
        int waves = (MTOT / 16) * (H2_ / 128);              // 8192
        gemm_bf16_blk<8><<<waves / 8, 256, 0, stream>>>(c0bf, fc2bt, z2,
                                                        MTOT, H2_, NHEADS * H1_);
    }
    // 5) attention on z2 -> c2
    gat_out<<<MTOT / 8, 256, 0, stream>>>(z2, neigh, c2);
    // 6) relu-max pool over nodes -> pooled (T x 128)
    pool_max<<<T_, H2_, 0, stream>>>(c2, pooled);
    // 7) sequential GRU + heads + SIR
    seq_tail<<<1, 256, 0, stream>>>(pooled, Wih, Whh, bih, bhh, r1W, r1b, r2W, r2b,
                                    hx0, Npop, I_, R_, S_, It, Rt, out);
}
